// Model_39676907884686
// MI455X (gfx1250) — matrix operand order repaired and checked by reading, not compiled, not run on hardware
//
#include <hip/hip_runtime.h>
#include <hip/hip_bf16.h>
#include <stdint.h>

// ---------------------------------------------------------------------------
// Flash attention fwd (B=4, Sq=Sk=4096, D=Dv=64, fp32 in/out) for gfx1250.
//  - Pre-pass: K -> bf16 [B][Sk][D]; V -> bf16, transposed AND column-permuted
//    per 64-tile: Vt[b][v][kt*64 + (j%16)*4 + j/16] = V[b][kt*64+j][v].
//  - Main: bf16 WMMA, fp32 accum, online softmax in log2 domain, DPP row_ror
//    butterflies for row max/sum, v_perm-packed bf16 P staging (b64 stores),
//    async double-buffered LDS tiles (GLOBAL_LOAD_ASYNC_TO_LDS_B128).
//  - 2-way Sk split + merge when workspace allows.
// ---------------------------------------------------------------------------

typedef __attribute__((ext_vector_type(16))) __bf16          v16bf;
typedef __attribute__((ext_vector_type(2)))  __bf16          v2bf;
typedef __attribute__((ext_vector_type(8)))  float           v8f;
typedef __attribute__((ext_vector_type(8)))  unsigned int    v8u;
typedef __attribute__((ext_vector_type(4)))  int             v4i;

#define B_   4
#define SQ_  4096
#define SK_  4096
#define D_   64
#define DV_  64
#define BM_  64
#define BN_  64
#define PAD_ 72      // padded LDS row -> 144B, 16B aligned, conflict-free
#define NWAVE_ 4
#define NT_  (SK_ / BN_)

#define KEEP8    205u         // byte threshold: keep prob ~0.8
#define INV_KEEP 1.25f
#define LOG2E    1.44269504089f

#if defined(__has_builtin)
#if __has_builtin(__builtin_amdgcn_global_load_async_to_lds_b128)
#define HAVE_ASYNC 1
#endif
#endif

__device__ __forceinline__ float fexp2(float x) {
#if defined(__has_builtin) && __has_builtin(__builtin_amdgcn_exp2f)
    return __builtin_amdgcn_exp2f(x);
#else
    return exp2f(x);
#endif
}

__device__ __forceinline__ unsigned short f2bf(float f) {
    union { float f; unsigned u; } x; x.f = f;
    unsigned r = x.u + 0x7FFFu + ((x.u >> 16) & 1u);
    return (unsigned short)(r >> 16);
}

// pack two floats -> bf16x2 dword: round both, then one v_perm_b32 grabs the
// two high halves. Avoids 16-bit half-register insert spam.
__device__ __forceinline__ unsigned pack_bf2(float a, float b) {
#if defined(__has_builtin) && __has_builtin(__builtin_amdgcn_cvt_pk_bf16_f32)
    v2bf p = __builtin_amdgcn_cvt_pk_bf16_f32(a, b);
    return __builtin_bit_cast(unsigned, p);
#else
    const unsigned au = __builtin_bit_cast(unsigned, a);
    const unsigned bu = __builtin_bit_cast(unsigned, b);
    const unsigned ra = au + 0x7FFFu + ((au >> 16) & 1u);
    const unsigned rb = bu + 0x7FFFu + ((bu >> 16) & 1u);
    // dst bytes: {rb[3],rb[2],ra[3],ra[2]}  (S0=rb -> bytes 7..4, S1=ra -> 3..0)
    return __builtin_amdgcn_perm(rb, ra, 0x07060302u);
#endif
}

// DPP16 row_ror rotate within 16-lane rows (pure VALU cross-lane)
#if defined(__has_builtin) && __has_builtin(__builtin_amdgcn_update_dpp)
#define HAVE_DPP 1
#endif
template <int CTRL>
__device__ __forceinline__ float dpp_rorf(float x) {
#ifdef HAVE_DPP
    const int v = __builtin_amdgcn_update_dpp(0, __builtin_bit_cast(int, x),
                                              CTRL, 0xF, 0xF, true);
    return __builtin_bit_cast(float, v);
#else
    const int off = (CTRL == 0x121) ? 1 : (CTRL == 0x122) ? 2 : (CTRL == 0x124) ? 4 : 8;
    return __shfl_xor(x, off, 16);
#endif
}
__device__ __forceinline__ float row_max16(float x) {
    x = fmaxf(x, dpp_rorf<0x121>(x));
    x = fmaxf(x, dpp_rorf<0x122>(x));
    x = fmaxf(x, dpp_rorf<0x124>(x));
    x = fmaxf(x, dpp_rorf<0x128>(x));
    return x;
}
__device__ __forceinline__ float row_sum16(float x) {
    x += dpp_rorf<0x121>(x);
    x += dpp_rorf<0x122>(x);
    x += dpp_rorf<0x124>(x);
    x += dpp_rorf<0x128>(x);
    return x;
}

__device__ __forceinline__ unsigned hash3(unsigned x, unsigned y, unsigned z) {
    unsigned h = x * 0x9E3779B1u + y * 0x85EBCA77u + z * 0xC2B2AE3Du + 42u * 0x27D4EB2Fu;
    h ^= h >> 16; h *= 0x7FEB352Du;
    h ^= h >> 15; h *= 0x846CA68Bu;
    h ^= h >> 16;
    return h;
}

__device__ __forceinline__ v16bf mk_bf_frag(uint4 lo, uint4 hi) {
    v8u t;
    t[0] = lo.x; t[1] = lo.y; t[2] = lo.z; t[3] = lo.w;
    t[4] = hi.x; t[5] = hi.y; t[6] = hi.z; t[7] = hi.w;
    return __builtin_bit_cast(v16bf, t);
}

__device__ __forceinline__ void cp16(const unsigned short* g, unsigned short* l) {
#ifdef HAVE_ASYNC
    typedef __attribute__((address_space(1))) v4i* gv4_t;
    typedef __attribute__((address_space(3))) v4i* lv4_t;
    gv4_t gp = (gv4_t)(const __attribute__((address_space(1))) void*)(const void*)g;
    lv4_t lp = (lv4_t)(__attribute__((address_space(3))) void*)(void*)l;
    __builtin_amdgcn_global_load_async_to_lds_b128(gp, lp, 0, 0);
#else
    *(uint4*)l = *(const uint4*)g;
#endif
}

__device__ __forceinline__ void wait_async0() {
#ifdef HAVE_ASYNC
#if __has_builtin(__builtin_amdgcn_s_wait_asynccnt)
    __builtin_amdgcn_s_wait_asynccnt(0);
#else
    asm volatile("s_wait_asynccnt 0" ::: "memory");
#endif
#endif
}

// ---------------------------------------------------------------------------
// Pre-pass: Kb[b][n][d] = bf16(K[b][n][d])
//           Vt[b][v][kt*64 + (j%16)*4 + j/16] = bf16(V[b][kt*64+j][v])
// ---------------------------------------------------------------------------
__global__ void cvt_kv(const float* __restrict__ K, const float* __restrict__ V,
                       unsigned short* __restrict__ Kb, unsigned short* __restrict__ Vt) {
    const size_t i = (size_t)blockIdx.x * blockDim.x + threadIdx.x;
    const size_t total4 = (size_t)B_ * SK_ * D_ / 4;
    if (i >= total4) return;

    const float4 kf = ((const float4*)K)[i];
    uint2 kp;
    kp.x = pack_bf2(kf.x, kf.y);
    kp.y = pack_bf2(kf.z, kf.w);
    ((uint2*)Kb)[i] = kp;

    const float4 vf = ((const float4*)V)[i];
    const size_t flat = i * 4;
    const int b  = (int)(flat / ((size_t)SK_ * DV_));
    const int r  = (int)(flat % ((size_t)SK_ * DV_));
    const int n  = r / DV_;                 // original key index
    const int v0 = r % DV_;
    const int nl  = n & (BN_ - 1);          // within 64-tile
    const int kt  = n >> 6;
    const int pos = (nl & 15) * 4 + (nl >> 4);          // permuted column
    const size_t col = (size_t)kt * BN_ + pos;
    Vt[((size_t)b * DV_ + v0 + 0) * SK_ + col] = f2bf(vf.x);
    Vt[((size_t)b * DV_ + v0 + 1) * SK_ + col] = f2bf(vf.y);
    Vt[((size_t)b * DV_ + v0 + 2) * SK_ + col] = f2bf(vf.z);
    Vt[((size_t)b * DV_ + v0 + 3) * SK_ + col] = f2bf(vf.w);
}

// ---------------------------------------------------------------------------
// Main flash-attention kernel (templated epilogue).
// ---------------------------------------------------------------------------
template <bool PARTIAL>
__global__ __launch_bounds__(128, 2)
void fattn_bf16_wmma(const float* __restrict__ Qf,
                     const unsigned short* __restrict__ Kb,
                     const unsigned short* __restrict__ Vt,
                     const int* __restrict__ inv_scale,
                     float* __restrict__ Out,
                     float2* __restrict__ Ml) {
    __shared__ __align__(16) unsigned short k_lds[2][BN_][PAD_];
    __shared__ __align__(16) unsigned short vt_lds[2][DV_][PAD_];
    __shared__ __align__(16) unsigned short p_lds[NWAVE_][16][PAD_];

    const int t    = threadIdx.x;
    const int lane = t & 31;
    const int wv   = t >> 5;
    const int half = lane >> 4;
    const int lm   = lane & 15;

    const int nbq = B_ * (SQ_ / BM_);
    const int bq  = blockIdx.x % nbq;
    const int sp  = PARTIAL ? (blockIdx.x / nbq) : 0;
    const int b   = bq / (SQ_ / BM_);
    const int q0  = (bq % (SQ_ / BM_)) * BM_;

    const int kt0   = PARTIAL ? sp * (NT_ / 2) : 0;
    const int ktend = PARTIAL ? kt0 + (NT_ / 2) : NT_;

    const float qscale = LOG2E / (float)(*inv_scale);
    const int qrow = q0 + wv * 16 + lm;

    // ---- Q A-fragments (16-bit A layout), scaled + perm-packed once ----
    v16bf a_q[2];
    {
        const float* qp = Qf + ((size_t)b * SQ_ + qrow) * D_;
#pragma unroll
        for (int dc = 0; dc < 2; ++dc) {
            v8u a;
            const int base = dc * 32 + half * 8;
#pragma unroll
            for (int e = 0; e < 4; ++e)
                a[e] = pack_bf2(qp[base + 2 * e] * qscale, qp[base + 2 * e + 1] * qscale);
#pragma unroll
            for (int e = 0; e < 4; ++e)
                a[4 + e] = pack_bf2(qp[base + 16 + 2 * e] * qscale,
                                    qp[base + 16 + 2 * e + 1] * qscale);
            a_q[dc] = __builtin_bit_cast(v16bf, a);
        }
    }

    auto stage = [&](int kt, int buf) {
        const size_t kgbase = ((size_t)b * SK_ + (size_t)kt * BN_) * D_;
        const size_t vgbase = (size_t)b * DV_ * SK_ + (size_t)kt * BN_;
#pragma unroll
        for (int i = 0; i < 4; ++i) {
            const int c  = t + i * 128;
            const int n  = c >> 3;
            const int e0 = (c & 7) * 8;
            cp16(Kb + kgbase + (size_t)n * D_ + e0, &k_lds[buf][n][e0]);
            cp16(Vt + vgbase + (size_t)n * SK_ + e0, &vt_lds[buf][n][e0]);
        }
    };

    v8f   acc[4];
    float mrow[8], lrow[8];
#pragma unroll
    for (int vc = 0; vc < 4; ++vc) acc[vc] = (v8f)0.0f;
#pragma unroll
    for (int r = 0; r < 8; ++r) { mrow[r] = -3.0e38f; lrow[r] = 0.0f; }

    stage(kt0, 0);

    for (int kt = kt0; kt < ktend; ++kt) {
        const int cur = (kt - kt0) & 1;
        const int k0  = kt * BN_;

        wait_async0();
        __syncthreads();

        if (kt + 1 < ktend) stage(kt + 1, cur ^ 1);

        // ---- S = Q K^T (scaled, log2 domain) ----
        v8f s[4];
#pragma unroll
        for (int nc = 0; nc < 4; ++nc) {
            v8f c = (v8f)0.0f;
#pragma unroll
            for (int dc = 0; dc < 2; ++dc) {
                const uint4* bp = (const uint4*)&k_lds[cur][nc * 16 + lm][dc * 32 + half * 8];
                const v16bf bb = mk_bf_frag(bp[0], bp[2]);
                c = __builtin_amdgcn_wmma_f32_16x16x32_bf16(
                        false, a_q[dc], false, bb, (short)0, c, false, false);
            }
            s[nc] = c;
        }

        // ---- Online softmax + dropout; P staged packed (permuted cols) ----
#pragma unroll
        for (int r = 0; r < 8; ++r) {
            const int   qr   = q0 + wv * 16 + r + half * 8;
            const float mOld = mrow[r];
            float tm = fmaxf(fmaxf(s[0][r], s[1][r]), fmaxf(s[2][r], s[3][r]));
            tm = row_max16(tm);
            const float mNew = fmaxf(mOld, tm);

            const unsigned h = hash3((unsigned)b, (unsigned)qr, (unsigned)(k0 + lm));

            float pj[4], pd[4];
            float rs = 0.0f;
#pragma unroll
            for (int nc = 0; nc < 4; ++nc) {
                pj[nc] = fexp2(s[nc][r] - mNew);
                rs += pj[nc];
                const bool keep = ((h >> (8 * nc)) & 0xFFu) < KEEP8;
                pd[nc] = keep ? pj[nc] * INV_KEEP : 0.0f;
            }
            // lane lm owns permuted columns lm*4 .. lm*4+3 -> one b64 store
            uint2 pk;
            pk.x = pack_bf2(pd[0], pd[1]);
            pk.y = pack_bf2(pd[2], pd[3]);
            *(uint2*)&p_lds[wv][r + half * 8][lm * 4] = pk;

            rs = row_sum16(rs);
            const float alpha = fexp2(mOld - mNew);
            lrow[r] = lrow[r] * alpha + rs;
            mrow[r] = mNew;
#pragma unroll
            for (int vc = 0; vc < 4; ++vc) acc[vc][r] *= alpha;
        }

        // ---- O += P * V (both sides share the per-tile column permutation) ----
#pragma unroll
        for (int kc = 0; kc < 2; ++kc) {
            const uint4* pp = (const uint4*)&p_lds[wv][lm][kc * 32 + half * 8];
            const v16bf a_p = mk_bf_frag(pp[0], pp[2]);
#pragma unroll
            for (int vc = 0; vc < 4; ++vc) {
                const uint4* bp = (const uint4*)&vt_lds[cur][vc * 16 + lm][kc * 32 + half * 8];
                const v16bf bb = mk_bf_frag(bp[0], bp[2]);
                acc[vc] = __builtin_amdgcn_wmma_f32_16x16x32_bf16(
                              false, a_p, false, bb, (short)0, acc[vc], false, false);
            }
        }
    }

    // ---- Epilogue ----
    if (PARTIAL) {
        float* Op = Out + (size_t)sp * B_ * SQ_ * DV_;
#pragma unroll
        for (int vc = 0; vc < 4; ++vc)
#pragma unroll
            for (int r = 0; r < 8; ++r) {
                const int qr = q0 + wv * 16 + r + half * 8;
                Op[((size_t)b * SQ_ + qr) * DV_ + vc * 16 + lm] = acc[vc][r];
            }
        if (lm == 0) {
#pragma unroll
            for (int r = 0; r < 8; ++r) {
                const int qr = q0 + wv * 16 + r + half * 8;
                Ml[(size_t)sp * B_ * SQ_ + (size_t)b * SQ_ + qr] =
                    make_float2(mrow[r], lrow[r]);
            }
        }
    } else {
#pragma unroll
        for (int vc = 0; vc < 4; ++vc)
#pragma unroll
            for (int r = 0; r < 8; ++r) {
                const int qr = q0 + wv * 16 + r + half * 8;
                Out[((size_t)b * SQ_ + qr) * DV_ + vc * 16 + lm] = acc[vc][r] / lrow[r];
            }
    }
}

// ---------------------------------------------------------------------------
// Merge the 2 Sk-splits
// ---------------------------------------------------------------------------
__global__ void merge2(const float* __restrict__ Op, const float2* __restrict__ Ml,
                       float* __restrict__ Out) {
    const size_t i = (size_t)blockIdx.x * blockDim.x + threadIdx.x;
    const size_t total = (size_t)B_ * SQ_ * DV_;
    if (i >= total) return;
    const size_t row = i / DV_;
    const float2 a = Ml[row];
    const float2 c = Ml[(size_t)B_ * SQ_ + row];
    const float m  = fmaxf(a.x, c.x);
    const float w0 = fexp2(a.x - m);
    const float w1 = fexp2(c.x - m);
    const float l  = a.y * w0 + c.y * w1;
    Out[i] = (Op[i] * w0 + Op[total + i] * w1) / l;
}

extern "C" void kernel_launch(void* const* d_in, const int* in_sizes, int n_in,
                              void* d_out, int out_size, void* d_ws, size_t ws_size,
                              hipStream_t stream) {
    (void)in_sizes; (void)n_in; (void)out_size;
    const float* Q = (const float*)d_in[0];
    const float* K = (const float*)d_in[1];
    const float* V = (const float*)d_in[2];
    const int*   s = (const int*)d_in[3];
    float* O = (float*)d_out;

    const size_t kv_elems = (size_t)B_ * SK_ * D_;
    unsigned short* Kb = (unsigned short*)d_ws;
    unsigned short* Vt = Kb + kv_elems;
    const size_t kv_bytes = 2 * kv_elems * sizeof(unsigned short);

    const size_t op_elems = (size_t)B_ * SQ_ * DV_;
    const size_t ml_elems = (size_t)B_ * SQ_;
    const size_t split_bytes = 2 * op_elems * sizeof(float) + 2 * ml_elems * sizeof(float2);

    {
        const size_t total4 = kv_elems / 4;
        cvt_kv<<<dim3((unsigned)((total4 + 255) / 256)), dim3(256), 0, stream>>>(K, V, Kb, Vt);
    }

    const int nbq = B_ * (SQ_ / BM_);   // 256
    if (ws_size >= kv_bytes + split_bytes) {
        float*  Op = (float*)((char*)d_ws + kv_bytes);
        float2* Ml = (float2*)(Op + 2 * op_elems);
        fattn_bf16_wmma<true><<<dim3(2 * nbq), dim3(128), 0, stream>>>(Q, Kb, Vt, s, Op, Ml);
        merge2<<<dim3((unsigned)((op_elems + 255) / 256)), dim3(256), 0, stream>>>(Op, Ml, O);
    } else {
        fattn_bf16_wmma<false><<<dim3(nbq), dim3(128), 0, stream>>>(Q, Kb, Vt, s, O, nullptr);
    }
}
